// WanSelfAttention_82729660055643
// MI455X (gfx1250) — matrix operand order repaired and checked by reading, not compiled, not run on hardware
//
#include <hip/hip_runtime.h>
#include <hip/hip_bf16.h>
#include <stdint.h>

typedef __attribute__((ext_vector_type(16))) _Float16 v16h;
typedef __attribute__((ext_vector_type(8)))  _Float16 v8h;
typedef __attribute__((ext_vector_type(8)))  float    v8f;
typedef __attribute__((ext_vector_type(4)))  _Float16 h4;
typedef __attribute__((ext_vector_type(4)))  float    f4;
typedef __attribute__((ext_vector_type(4)))  int      v4i;

typedef __attribute__((address_space(1))) v4i* gv4i_p;   // global v4i*
typedef __attribute__((address_space(3))) v4i* lv4i_p;   // LDS v4i*

#if __has_builtin(__builtin_amdgcn_global_load_async_to_lds_b128)
#define HAS_ASYNC_LDS 1
#else
#define HAS_ASYNC_LDS 0
#endif

namespace {

constexpr int Fd = 8, Hd = 16, Wd = 24;
constexpr int S   = Fd * Hd * Wd;      // 3072
constexpr int DIM = 1536, NH = 12, Dh = 128, Ch = 64, C0 = 22, C1 = 21;
constexpr int LC  = 2 * Hd * Wd;       // 768 cached positions
constexpr int LK  = LC + S;            // 3840 total kv length
constexpr float EPSF = 1e-6f;

__device__ inline v8f wmma16(v16h a, v16h b, v8f c) {
  return __builtin_amdgcn_wmma_f32_16x16x32_f16(false, a, false, b, (short)0, c,
                                                false, false);
}

// 16-byte global -> LDS copy; async (ASYNCcnt) when available.
__device__ inline void copy16(const _Float16* g, _Float16* l) {
#if HAS_ASYNC_LDS
  __builtin_amdgcn_global_load_async_to_lds_b128((gv4i_p)g, (lv4i_p)l, 0, 0);
#else
  *(v8h*)l = *(const v8h*)g;
#endif
}

#if HAS_ASYNC_LDS
#define WAIT_ASYNC(n) asm volatile("s_wait_asynccnt " #n ::: "memory")
#else
#define WAIT_ASYNC(n) asm volatile("" ::: "memory")
#endif

// A-fragment (16x32 f16, MxK): lane l -> row m = l&15, k = (l>>4)*16 + h.
__device__ inline v16h frag_a(const _Float16* p, int ld, int lane) {
  const _Float16* r = p + (size_t)(lane & 15) * ld + ((lane >> 4) << 4);
  v16h a;
#pragma unroll
  for (int h = 0; h < 8; ++h) a[h] = r[h];
#pragma unroll
  for (int h = 0; h < 8; ++h) a[8 + h] = r[8 + h];
  return a;
}

// B-fragment (32x16 f16, KxN): lane l -> col n = l&15, k = (l>>4)*16 + h.
// Source must be N-major with contiguous K (stride ld per column).
__device__ inline v16h frag_b(const _Float16* p, int ld, int lane) {
  const _Float16* r = p + (size_t)(lane & 15) * ld + ((lane >> 4) << 4);
  v16h b;
#pragma unroll
  for (int h = 0; h < 16; ++h) b[h] = r[h];
  return b;
}

// ---------------------------------------------------------------------------
// convert_f16: elementwise f32 -> f16 (x prep).
// ---------------------------------------------------------------------------
__global__ __launch_bounds__(256)
void convert_f16_kernel(const float* __restrict__ src, _Float16* __restrict__ dst,
                        int n4) {
  int i = blockIdx.x * 256 + threadIdx.x;
  if (i >= n4) return;
  f4 v = *(const f4*)(src + (size_t)i * 4);
  h4 h;
  h[0] = (_Float16)v[0]; h[1] = (_Float16)v[1];
  h[2] = (_Float16)v[2]; h[3] = (_Float16)v[3];
  *(h4*)(dst + (size_t)i * 4) = h;
}

// ---------------------------------------------------------------------------
// transpose_f16: W[k][n] f32 -> Wt[n][k] f16, 32x32 LDS tiles.
// ---------------------------------------------------------------------------
__global__ __launch_bounds__(256)
void transpose_f16_kernel(const float* __restrict__ Wsrc,
                          _Float16* __restrict__ Wdst, int K, int N) {
  __shared__ float tile[32][33];
  const int k0 = blockIdx.x * 32, n0 = blockIdx.y * 32;
  const int tx = threadIdx.x & 31, ty = threadIdx.x >> 5;  // 32 x 8
#pragma unroll
  for (int i = 0; i < 32; i += 8)
    tile[ty + i][tx] = Wsrc[(size_t)(k0 + ty + i) * N + n0 + tx];
  __syncthreads();
#pragma unroll
  for (int i = 0; i < 32; i += 8)
    Wdst[(size_t)(n0 + ty + i) * K + k0 + tx] = (_Float16)tile[tx][ty + i];
}

// ---------------------------------------------------------------------------
// GEMM: Co(MxN,f32) = Ah(MxK,f16) @ Bt(NxK,f16 N-major)^T' + bias.
// Block tile 128x128, BK=32, 8 waves, wave tile 16x128 (8 WMMA / K-step).
// Double-buffered LDS fed by async global->LDS copies; steady-state loop is
// straight-line: stage(next) -> s_wait_asynccnt 4 -> barrier -> 8 WMMA.
// ---------------------------------------------------------------------------
__global__ __launch_bounds__(256)
void gemm_f16_kernel(const _Float16* __restrict__ Ah, const _Float16* __restrict__ Bt,
                     const float* __restrict__ bias, float* __restrict__ Co,
                     int M, int N, int K) {
  constexpr int BM = 128, BN = 128, BK = 32, LDA = BK + 8, LDB = BK + 8;
  __shared__ __align__(16) _Float16 Al[2][BM * LDA];
  __shared__ __align__(16) _Float16 Bl[2][BN * LDB];
  const int t = threadIdx.x;
  const int wave = t >> 5, lane = t & 31;
  const int m0 = blockIdx.x * BM, n0 = blockIdx.y * BN;

  v8f acc[8];
#pragma unroll
  for (int j = 0; j < 8; ++j)
#pragma unroll
    for (int r = 0; r < 8; ++r) acc[j][r] = 0.f;

  // 128 rows x 4 chunks(16B) = 512 chunks per tile; 2 per thread per matrix.
  auto stage = [&](int k0, int buf) {
#pragma unroll
    for (int i = 0; i < 2; ++i) {
      int e = t + (i << 8);
      int row = e >> 2, c8 = (e & 3) << 3;
      copy16(Ah + (size_t)(m0 + row) * K + k0 + c8, &Al[buf][row * LDA + c8]);
      copy16(Bt + (size_t)(n0 + row) * K + k0 + c8, &Bl[buf][row * LDB + c8]);
    }
  };

  // Batch-load all fragments, then issue the WMMA chain.
  auto compute = [&](int buf) {
    v16h a = frag_a(&Al[buf][(wave << 4) * LDA], LDA, lane);
    v16h b[8];
#pragma unroll
    for (int j = 0; j < 8; ++j) b[j] = frag_b(&Bl[buf][(j << 4) * LDB], LDB, lane);
#pragma unroll
    for (int j = 0; j < 8; ++j) acc[j] = wmma16(a, b[j], acc[j]);
  };

  const int nk = K >> 5;
  stage(0, 0);
  for (int ki = 0; ki < nk - 1; ++ki) {
    const int buf = ki & 1;
    stage((ki + 1) << 5, buf ^ 1);
    WAIT_ASYNC(4);   // previous tile's 4 copies/thread done; next 4 in flight
    __syncthreads();
    compute(buf);
    __syncthreads();
  }
  WAIT_ASYNC(0);
  __syncthreads();
  compute((nk - 1) & 1);

  // C-layout: lane l, vgpr r -> row m = r + (l>>4)*8, col n = l&15.
  const int mr = m0 + (wave << 4) + ((lane >> 4) << 3);
#pragma unroll
  for (int j = 0; j < 8; ++j) {
    int nc = n0 + (j << 4) + (lane & 15);
    float bb = bias[nc];
#pragma unroll
    for (int r = 0; r < 8; ++r)
      Co[(size_t)(mr + r) * N + nc] = acc[j][r] + bb;
  }
}

// ---------------------------------------------------------------------------
// pack_qkv: one block per position s. RMSNorm(q,k)*w, 3D RoPE (shift 0):
//   Qh[head][s][d], Kh[head][LC+s][d], Vt[head][d][LC+s] (transposed).
// ---------------------------------------------------------------------------
__global__ __launch_bounds__(256)
void pack_qkv_kernel(const float* __restrict__ qr, const float* __restrict__ kr,
                     const float* __restrict__ vr, const float* __restrict__ wqn,
                     const float* __restrict__ wkn, const float* __restrict__ fcos,
                     const float* __restrict__ fsin, _Float16* __restrict__ Qh,
                     _Float16* __restrict__ Kh, _Float16* __restrict__ Vt) {
  const int s = blockIdx.x, t = threadIdx.x;
  const float* q = qr + (size_t)s * DIM;
  const float* k = kr + (size_t)s * DIM;
  const float* v = vr + (size_t)s * DIM;

  __shared__ float red[2][256];
  float aq = 0.f, ak = 0.f;
  for (int i = t; i < DIM; i += 256) {
    float x = q[i]; aq += x * x;
    float y = k[i]; ak += y * y;
  }
  red[0][t] = aq; red[1][t] = ak;
  __syncthreads();
  for (int off = 128; off; off >>= 1) {
    if (t < off) { red[0][t] += red[0][t + off]; red[1][t] += red[1][t + off]; }
    __syncthreads();
  }
  const float rq = rsqrtf(red[0][0] / DIM + EPSF);
  const float rk = rsqrtf(red[1][0] / DIM + EPSF);

  const int f = s / (Hd * Wd);
  const int rem = s - f * Hd * Wd;
  const int hh = rem / Wd, ww = rem - hh * Wd;

  for (int p = t; p < DIM / 2; p += 256) {
    int dg = 2 * p;
    int n = dg >> 7, dd = dg & 127, c = dd >> 1;
    int row = (c < C0) ? f : ((c < C0 + C1) ? hh : ww);
    float fc = fcos[row * Ch + c], fs = fsin[row * Ch + c];

    float qa = q[dg] * rq * wqn[dg], qb = q[dg + 1] * rq * wqn[dg + 1];
    float ka = k[dg] * rk * wkn[dg], kb = k[dg + 1] * rk * wkn[dg + 1];

    size_t qo = ((size_t)n * S + s) * Dh + dd;
    Qh[qo]     = (_Float16)(qa * fc - qb * fs);
    Qh[qo + 1] = (_Float16)(qa * fs + qb * fc);

    size_t ko = ((size_t)n * LK + (LC + s)) * Dh + dd;
    Kh[ko]     = (_Float16)(ka * fc - kb * fs);
    Kh[ko + 1] = (_Float16)(ka * fs + kb * fc);

    size_t vo = ((size_t)(n * Dh + dd)) * LK + (LC + s);
    Vt[vo]      = (_Float16)v[dg];
    Vt[vo + LK] = (_Float16)v[dg + 1];
  }
}

// ---------------------------------------------------------------------------
// pack_prev: cached K/V -> shifted RoPE * head_mask -> f16 caches.
// prev_k/v layout: (LC, NH, Dh).
// ---------------------------------------------------------------------------
__global__ __launch_bounds__(256)
void pack_prev_kernel(const float* __restrict__ pk, const float* __restrict__ pv,
                      const unsigned char* __restrict__ hm,
                      const float* __restrict__ fcos, const float* __restrict__ fsin,
                      const int* __restrict__ shiftp,
                      _Float16* __restrict__ Kh, _Float16* __restrict__ Vt) {
  int idx = blockIdx.x * 256 + threadIdx.x;
  const int total = LC * NH * (Dh / 2);
  if (idx >= total) return;
  const int shift = *shiftp;
  int c = idx % (Dh / 2);
  int tmp = idx / (Dh / 2);
  int n = tmp % NH;
  int l = tmp / NH;
  int dd = 2 * c;
  int f = l / (Hd * Wd);
  int rem = l - f * Hd * Wd;
  int hh = rem / Wd, ww = rem - hh * Wd;
  int row = (c < C0) ? (shift + f) : ((c < C0 + C1) ? hh : ww);
  float fc = fcos[row * Ch + c], fs = fsin[row * Ch + c];
  float m = hm[n] ? 1.f : 0.f;

  size_t io = (((size_t)l * NH) + n) * Dh + dd;
  float a = pk[io], b = pk[io + 1];
  size_t ko = ((size_t)n * LK + l) * Dh + dd;
  Kh[ko]     = (_Float16)((a * fc - b * fs) * m);
  Kh[ko + 1] = (_Float16)((a * fs + b * fc) * m);
  size_t vo = ((size_t)(n * Dh + dd)) * LK + l;
  Vt[vo]      = (_Float16)(pv[io] * m);
  Vt[vo + LK] = (_Float16)(pv[io + 1] * m);
}

// ---------------------------------------------------------------------------
// Flash attention: block = (128 q rows, 1 head); each wave owns 16 q rows.
// 32-key blocks: 8 score WMMAs + online softmax + 8 PV WMMAs.
// K/V fragments batch-loaded from L2 before each WMMA chain.
// Output written directly as f16 [s][DIM] for the O-projection GEMM.
// ---------------------------------------------------------------------------
__global__ __launch_bounds__(256)
void attn_kernel(const _Float16* __restrict__ Qh, const _Float16* __restrict__ Kh,
                 const _Float16* __restrict__ Vt, _Float16* __restrict__ Oh) {
  constexpr int LDP = 40;
  __shared__ __align__(16) _Float16 Plds[8 * 16 * LDP];
  const int t = threadIdx.x, wave = t >> 5, lane = t & 31;
  const int head = blockIdx.y;
  const int q0 = blockIdx.x * 128 + (wave << 4);
  const float scale = 0.088388347648318447f;  // 1/sqrt(128)

  const _Float16* Qp = Qh + ((size_t)head * S + q0) * Dh;
  v16h qa[4];
#pragma unroll
  for (int kk = 0; kk < 4; ++kk) qa[kk] = frag_a(Qp + kk * 32, Dh, lane);

  v8f O[8];
#pragma unroll
  for (int j = 0; j < 8; ++j)
#pragma unroll
    for (int r = 0; r < 8; ++r) O[j][r] = 0.f;
  float mrow[8], lrow[8];
#pragma unroll
  for (int r = 0; r < 8; ++r) { mrow[r] = -1e30f; lrow[r] = 0.f; }

  _Float16* Pw = &Plds[wave * 16 * LDP];
  const int mi = (lane >> 4) << 3;
  const int kc = lane & 15;

  for (int kb0 = 0; kb0 < LK; kb0 += 32) {
    const _Float16* Kp = Kh + ((size_t)head * LK + kb0) * Dh;
    // Batch-load 8 K fragments (two 16-key column tiles x 4 d-chunks).
    v16h kb[8];
#pragma unroll
    for (int kk = 0; kk < 4; ++kk) {
      kb[kk]     = frag_b(Kp + kk * 32, Dh, lane);
      kb[4 + kk] = frag_b(Kp + 16 * Dh + kk * 32, Dh, lane);
    }
    v8f s0, s1;
#pragma unroll
    for (int r = 0; r < 8; ++r) { s0[r] = 0.f; s1[r] = 0.f; }
#pragma unroll
    for (int kk = 0; kk < 4; ++kk) {
      s0 = wmma16(qa[kk], kb[kk], s0);
      s1 = wmma16(qa[kk], kb[4 + kk], s1);
    }
    // Online softmax; rows striped across the 16-lane half-wave.
    float alpha[8];
#pragma unroll
    for (int r = 0; r < 8; ++r) {
      float x0 = s0[r] * scale, x1 = s1[r] * scale;
      float vmx = fmaxf(x0, x1);
#pragma unroll
      for (int off = 8; off; off >>= 1) vmx = fmaxf(vmx, __shfl_xor(vmx, off, 32));
      float mn = fmaxf(mrow[r], vmx);
      alpha[r] = __expf(mrow[r] - mn);
      mrow[r] = mn;
      float p0 = __expf(x0 - mn), p1 = __expf(x1 - mn);
      Pw[(mi + r) * LDP + kc]      = (_Float16)p0;
      Pw[(mi + r) * LDP + 16 + kc] = (_Float16)p1;
      float ps = p0 + p1;
#pragma unroll
      for (int off = 8; off; off >>= 1) ps += __shfl_xor(ps, off, 32);
      lrow[r] = lrow[r] * alpha[r] + ps;
    }
#pragma unroll
    for (int j = 0; j < 8; ++j)
#pragma unroll
      for (int r = 0; r < 8; ++r) O[j][r] *= alpha[r];

    asm volatile("s_wait_dscnt 0x0" ::: "memory");  // P visible to frag_a
    v16h pa = frag_a(Pw, LDP, lane);
    const _Float16* Vp = Vt + (size_t)head * Dh * LK + kb0;
    v16h bv[8];
#pragma unroll
    for (int j = 0; j < 8; ++j)
      bv[j] = frag_b(Vp + (size_t)(j << 4) * LK, LK, lane);
#pragma unroll
    for (int j = 0; j < 8; ++j) O[j] = wmma16(pa, bv[j], O[j]);
  }

  float inv[8];
#pragma unroll
  for (int r = 0; r < 8; ++r) inv[r] = 1.f / lrow[r];
#pragma unroll
  for (int j = 0; j < 8; ++j) {
    int col = head * Dh + (j << 4) + kc;
#pragma unroll
    for (int r = 0; r < 8; ++r)
      Oh[(size_t)(q0 + mi + r) * DIM + col] = (_Float16)(O[j][r] * inv[r]);
  }
}

}  // namespace

extern "C" void kernel_launch(void* const* d_in, const int* in_sizes, int n_in,
                              void* d_out, int out_size, void* d_ws, size_t ws_size,
                              hipStream_t stream) {
  (void)in_sizes; (void)n_in; (void)out_size; (void)ws_size;
  const float* x      = (const float*)d_in[0];
  const float* Wq     = (const float*)d_in[1];
  const float* bq     = (const float*)d_in[2];
  const float* Wk     = (const float*)d_in[3];
  const float* bk     = (const float*)d_in[4];
  const float* Wv     = (const float*)d_in[5];
  const float* bv     = (const float*)d_in[6];
  const float* Wo     = (const float*)d_in[7];
  const float* bo     = (const float*)d_in[8];
  const float* wqn    = (const float*)d_in[9];
  const float* wkn    = (const float*)d_in[10];
  const float* prev_k = (const float*)d_in[11];
  const float* prev_v = (const float*)d_in[12];
  const float* fcos   = (const float*)d_in[13];
  const float* fsin   = (const float*)d_in[14];
  const unsigned char* hmask = (const unsigned char*)d_in[15];
  const int* shiftp   = (const int*)d_in[18];
  float* out = (float*)d_out;

  char* ws = (char*)d_ws;
  const size_t szRaw = (size_t)S * DIM * sizeof(float);
  const size_t szXh  = (size_t)S * DIM * sizeof(_Float16);
  const size_t szWt  = (size_t)DIM * DIM * sizeof(_Float16);
  const size_t szQ   = (size_t)NH * S * Dh * sizeof(_Float16);
  const size_t szK   = (size_t)NH * LK * Dh * sizeof(_Float16);
  float*    q_raw = (float*)(ws);
  float*    k_raw = (float*)(ws + szRaw);
  float*    v_raw = (float*)(ws + 2 * szRaw);
  _Float16* xh    = (_Float16*)(ws + 3 * szRaw);          // also attn output
  _Float16* Wt    = (_Float16*)(ws + 3 * szRaw + szXh);   // reused per weight
  _Float16* Qh    = (_Float16*)(ws + 3 * szRaw + szXh + szWt);
  _Float16* Kh    = (_Float16*)(ws + 3 * szRaw + szXh + szWt + szQ);
  _Float16* Vt    = (_Float16*)(ws + 3 * szRaw + szXh + szWt + szQ + szK);
  _Float16* oh    = xh;  // xh free after the 3 QKV GEMMs

  const dim3 gg(S / 128, DIM / 128);        // (24, 12)
  const dim3 gt(DIM / 32, DIM / 32);        // (48, 48)
  const int  nconv = (S * DIM / 4 + 255) / 256;

  convert_f16_kernel<<<nconv, 256, 0, stream>>>(x, xh, S * DIM / 4);

  transpose_f16_kernel<<<gt, 256, 0, stream>>>(Wq, Wt, DIM, DIM);
  gemm_f16_kernel<<<gg, 256, 0, stream>>>(xh, Wt, bq, q_raw, S, DIM, DIM);
  transpose_f16_kernel<<<gt, 256, 0, stream>>>(Wk, Wt, DIM, DIM);
  gemm_f16_kernel<<<gg, 256, 0, stream>>>(xh, Wt, bk, k_raw, S, DIM, DIM);
  transpose_f16_kernel<<<gt, 256, 0, stream>>>(Wv, Wt, DIM, DIM);
  gemm_f16_kernel<<<gg, 256, 0, stream>>>(xh, Wt, bv, v_raw, S, DIM, DIM);

  pack_qkv_kernel<<<S, 256, 0, stream>>>(q_raw, k_raw, v_raw, wqn, wkn,
                                         fcos, fsin, Qh, Kh, Vt);
  const int prevTot = LC * NH * (Dh / 2);
  pack_prev_kernel<<<(prevTot + 255) / 256, 256, 0, stream>>>(
      prev_k, prev_v, hmask, fcos, fsin, shiftp, Kh, Vt);

  transpose_f16_kernel<<<gt, 256, 0, stream>>>(Wo, Wt, DIM, DIM);
  attn_kernel<<<dim3(S / 128, NH), 256, 0, stream>>>(Qh, Kh, Vt, oh);

  gemm_f16_kernel<<<gg, 256, 0, stream>>>(oh, Wt, bo, out, S, DIM, DIM);
}
